// Network_47107201303209
// MI455X (gfx1250) — matrix operand order repaired and checked by reading, not compiled, not run on hardware
//
#include <hip/hip_runtime.h>
#include <hip/hip_bf16.h>

// ---- CDNA5 WMMA vector types ----
typedef __attribute__((ext_vector_type(16))) __bf16 v16bf;
typedef __attribute__((ext_vector_type(8)))  __bf16 v8bf;
typedef __attribute__((ext_vector_type(8)))  float  v8f;

// ---- SLAYER constants (from reference config) ----
#define THETA 10.0f
constexpr int T_N   = 500;  // time bins
constexpr int T_PAD = 512;  // padded time bins in bf16 buffers (zero tail)
constexpr int KSRM  = 77;   // SRM alpha kernel length (tau=10, eps=0.01)
constexpr int KREF  = 11;   // refractory kernel length (tau=1, eps=0.01, mult=-20)

__device__ __forceinline__ __bf16 u16_as_bf16(unsigned short u) {
  union { unsigned short s; __bf16 b; } c; c.s = u; return c.b;
}

// CDNA5 async global->LDS copy, GVS mode: mem = SADDR(base) + VADDR(u32) ; LDS[VDST(u32)]
__device__ __forceinline__ void async_ld_b128(unsigned lds_off,
                                              unsigned long long base,
                                              unsigned goff) {
  asm volatile("global_load_async_to_lds_b128 %0, %1, %2"
               :: "v"(lds_off), "v"(goff), "s"(base) : "memory");
}
__device__ __forceinline__ void wait_async0() {
  asm volatile("s_wait_asynccnt 0x0" ::: "memory");
}

// =====================================================================
// One-time packers (outside GEMM hot loop)
// =====================================================================
// W [Mrows][K] f32 -> hi/lo bf16 [Mpad][K] (split-bf16; rows >= Mrows zero)
__global__ __launch_bounds__(256) void snn_pack_w(const float* __restrict__ W,
                                                  __bf16* __restrict__ hi,
                                                  __bf16* __restrict__ lo,
                                                  int Mrows, int Mpad, int K)
{
  int idx = blockIdx.x * 256 + threadIdx.x;       // over Mpad*K (row-major)
  if (idx >= Mpad * K) return;
  int m = idx / K;
  float w = (m < Mrows) ? W[idx] : 0.f;           // idx == m*K+k valid when m<Mrows
  __bf16 h = (__bf16)w;
  hi[idx] = h;
  lo[idx] = (__bf16)(w - (float)h);
}

// x f32 [rows][T] -> bf16 [rows][Tpad] with zeroed tail (spikes are exact in bf16)
__global__ __launch_bounds__(256) void snn_pack_x(const float* __restrict__ x,
                                                  __bf16* __restrict__ xb,
                                                  int T, int Tpad, long total)
{
  long idx = (long)blockIdx.x * 256 + threadIdx.x;  // over rows*Tpad
  if (idx >= total) return;
  long r = idx / Tpad;
  int  t = (int)(idx - r * Tpad);
  xb[idx] = (t < T) ? (__bf16)x[r * T + t] : (__bf16)0.f;
}

// =====================================================================
// GEMM: U[b,m,t] = sum_k W[m,k] * X[b,k,t]  (bf16 WMMA, split hi/lo)
//  Block = 256 threads = 8 waves; each wave owns one 16x16 C tile.
//  K-tile = 64, double-buffered LDS: tile i+1's A copies are issued as
//  async global->LDS (ASYNCcnt) while tile i's 4 WMMAs run; one
//  s_wait_asynccnt + one barrier per K-tile.
// =====================================================================
template<int WM, int WN>
__global__ __launch_bounds__(256) void snn_wmma_gemm(
    const __bf16* __restrict__ Whi, const __bf16* __restrict__ Wlo,
    const __bf16* __restrict__ Xb, float* __restrict__ U,
    int Mrows, int Kdim, int T, int Tpad, int numMBlocks)
{
  constexpr int MB = WM * 16, NB = WN * 16;
  constexpr int KT = 64;                 // K per staging round
  constexpr int ACH = MB * (KT * 2 / 16);   // 16B chunks per A array per tile
  __shared__ __align__(16) __bf16 Ah[2][MB][KT];   // [buf][m][k]
  __shared__ __align__(16) __bf16 Al[2][MB][KT];
  __shared__ __align__(16) __bf16 Bs[2][NB][KT];   // [buf][t][k] (transposed)

  const int b     = blockIdx.y;
  const int mBlk  = blockIdx.x % numMBlocks;
  const int tBlk  = blockIdx.x / numMBlocks;
  const int mBase = mBlk * MB;
  const int t0    = tBlk * NB;
  const int tid   = threadIdx.x;
  const int wave  = tid >> 5;
  const int lane  = tid & 31;
  const int wm    = wave % WM;
  const int wn    = wave / WM;
  const int r     = lane & 15;
  const int h     = lane >> 4;

  const unsigned long long whiB = (unsigned long long)Whi;
  const unsigned long long wloB = (unsigned long long)Wlo;

  // A tile: MB rows x 128B, copied as 16B chunks via async DMA to LDS
  auto stageA = [&](int buf, int k0) {
    unsigned ldsAh = (unsigned)(size_t)&Ah[buf][0][0];
    unsigned ldsAl = (unsigned)(size_t)&Al[buf][0][0];
    for (int c = tid; c < ACH; c += 256) {
      int rowi = c >> 3, piece = c & 7;
      unsigned goff = (unsigned)(((mBase + rowi) * Kdim + k0) * 2 + piece * 16);
      async_ld_b128(ldsAh + (unsigned)c * 16, whiB, goff);
      async_ld_b128(ldsAl + (unsigned)c * 16, wloB, goff);
    }
  };
  // B tile: transpose [k][t] -> [t][k]; one u32 (= 2 t's) per element pair
  auto stageB = [&](int buf, int k0) {
    for (int q = tid; q < NB * (KT / 2); q += 256) {
      int kk = q / (NB / 2);
      int j  = q % (NB / 2);
      unsigned v = *(const unsigned*)&Xb[((long)(b * Kdim + k0 + kk)) * Tpad + t0 + 2 * j];
      Bs[buf][2 * j][kk]     = u16_as_bf16((unsigned short)(v & 0xffffu));
      Bs[buf][2 * j + 1][kk] = u16_as_bf16((unsigned short)(v >> 16));
    }
  };

  v8f acc = {0.f, 0.f, 0.f, 0.f, 0.f, 0.f, 0.f, 0.f};

  stageA(0, 0);
  stageB(0, 0);
  wait_async0();
  __syncthreads();

  const int arow = wm * 16 + r;
  const int bcol = wn * 16 + r;
  int cur = 0;
  for (int k0 = 0; k0 < Kdim; k0 += KT) {
    const int nk = k0 + KT;
    if (nk < Kdim) {            // prefetch next tile while computing this one
      stageA(cur ^ 1, nk);
      stageB(cur ^ 1, nk);
    }
#pragma unroll
    for (int s = 0; s < 2; ++s) {   // two K=32 WMMA sub-steps per tile
      // A 16x32: row M=r; elems 0..7 -> K=h*8+e ; elems 8..15 -> K=16+h*8+e
      v8bf aH0 = *(const v8bf*)&Ah[cur][arow][s * 32 + h * 8];
      v8bf aH1 = *(const v8bf*)&Ah[cur][arow][s * 32 + 16 + h * 8];
      v8bf aL0 = *(const v8bf*)&Al[cur][arow][s * 32 + h * 8];
      v8bf aL1 = *(const v8bf*)&Al[cur][arow][s * 32 + 16 + h * 8];
      // B 32x16: col N=r; elems 0..7 -> K=h*8+e ; elems 8..15 -> K=16+h*8+e
      v8bf b0 = *(const v8bf*)&Bs[cur][bcol][s * 32 + h * 8];
      v8bf b1 = *(const v8bf*)&Bs[cur][bcol][s * 32 + 16 + h * 8];

      v16bf aH, aL, bb;
#pragma unroll
      for (int e = 0; e < 8; ++e) {
        aH[e] = aH0[e]; aH[8 + e] = aH1[e];
        aL[e] = aL0[e]; aL[8 + e] = aL1[e];
        bb[e] = b0[e];  bb[8 + e] = b1[e];
      }
      // D = A_hi*B + (A_lo*B + C); bf16*bf16 products are exact in f32
      acc = __builtin_amdgcn_wmma_f32_16x16x32_bf16(false, aL, false, bb, (short)0, acc, false, false);
      acc = __builtin_amdgcn_wmma_f32_16x16x32_bf16(false, aH, false, bb, (short)0, acc, false, false);
    }
    wait_async0();      // next tile's async copies landed in LDS
    __syncthreads();    // publish next tile / retire reads of current
    cur ^= 1;
  }

  // C/D layout: VGPR j -> M=j (lanes 0-15) / j+8 (lanes 16-31), N = r
  const int col = t0 + wn * 16 + r;
#pragma unroll
  for (int j = 0; j < 8; ++j) {
    int m = mBase + wm * 16 + ((lane < 16) ? j : (j + 8));
    if (m < Mrows && col < T)
      U[(b * Mrows + m) * T + col] = acc[j];
  }
}

// =====================================================================
// PSP: causal FIR with SRM alpha kernel, one block per (b,n) row.
// =====================================================================
__global__ __launch_bounds__(512) void snn_psp(const float* __restrict__ u,
                                               float* __restrict__ y, int T)
{
  __shared__ float sdat[KSRM - 1 + T_N];
  __shared__ float kern[KSRM];
  const long row = blockIdx.x;
  const int  tid = threadIdx.x;
  for (int i = tid; i < KSRM - 1; i += blockDim.x) sdat[i] = 0.f;
  for (int i = tid; i < T; i += blockDim.x) sdat[KSRM - 1 + i] = u[row * T + i];
  if (tid < KSRM) kern[tid] = (tid * 0.1f) * __expf(1.0f - tid * 0.1f);
  __syncthreads();
  for (int t = tid; t < T; t += blockDim.x) {
    float acc = 0.f;
#pragma unroll 7
    for (int j = 0; j < KSRM; ++j)
      acc += kern[j] * sdat[KSRM - 1 + t - j];
    y[row * T + t] = acc;   // * TS (== 1.0)
  }
}

// =====================================================================
// Spike: sequential LIF threshold + refractory ring buffer.
// 256 neurons/block; T staged through LDS in 64-wide chunks
// (row stride 65 floats -> conflict-free per-lane scan).
// OutT = __bf16 (feeds next GEMM, stride T_PAD) or float (final output).
// =====================================================================
constexpr int TC = 64;
template<typename OutT>
__global__ __launch_bounds__(256) void snn_spike(const float* __restrict__ y,
                                                 OutT* __restrict__ s,
                                                 int totalRows, int T, int outStride)
{
  __shared__ float tile[256][TC + 1];
  const int tid     = threadIdx.x;
  const int rowBase = blockIdx.x * 256;
  const int myRow   = rowBase + tid;

  // refk[j] = REF_KERNEL[j+1] = -2*theta*(j+1)*e^{1-(j+1)}   (REF_KERNEL[0]==0)
  float refk[KREF - 1];
#pragma unroll
  for (int j = 0; j < KREF - 1; ++j) {
    float k = (float)(j + 1);
    refk[j] = -2.0f * THETA * k * __expf(1.0f - k);
  }
  float buf[KREF];
#pragma unroll
  for (int j = 0; j < KREF; ++j) buf[j] = 0.f;

  for (int t0 = 0; t0 < T; t0 += TC) {
    const int chunk = min(TC, T - t0);
    __syncthreads();
    for (int idx = tid; idx < 256 * chunk; idx += 256) {   // coalesced along t
      int rr = idx / chunk, tt = idx % chunk;
      int row = rowBase + rr;
      tile[rr][tt] = (row < totalRows) ? y[(long)row * T + t0 + tt] : 0.f;
    }
    __syncthreads();
    if (myRow < totalRows) {
      for (int tt = 0; tt < chunk; ++tt) {
        float u  = tile[tid][tt] + buf[0];
        float sp = (u >= THETA) ? 1.0f : 0.0f;   // amplitude 1/TS = 1
#pragma unroll
        for (int j = 0; j < KREF - 1; ++j) buf[j] = buf[j + 1];
        buf[KREF - 1] = 0.f;
#pragma unroll
        for (int j = 0; j < KREF - 1; ++j) buf[j] += sp * refk[j];
        tile[tid][tt] = sp;
      }
    }
    __syncthreads();
    for (int idx = tid; idx < 256 * chunk; idx += 256) {
      int rr = idx / chunk, tt = idx % chunk;
      int row = rowBase + rr;
      if (row < totalRows) s[(long)row * outStride + t0 + tt] = (OutT)tile[rr][tt];
    }
  }
  // zero padded tail (outStride > T)
  const int pad = outStride - T;
  for (int idx = tid; idx < 256 * pad; idx += 256) {
    int rr = idx / pad, tt = idx % pad;
    int row = rowBase + rr;
    if (row < totalRows) s[(long)row * outStride + T + tt] = (OutT)0.f;
  }
}

// =====================================================================
extern "C" void kernel_launch(void* const* d_in, const int* in_sizes, int n_in,
                              void* d_out, int out_size, void* d_ws, size_t ws_size,
                              hipStream_t stream)
{
  (void)in_sizes; (void)n_in; (void)out_size; (void)ws_size;
  const float* x  = (const float*)d_in[0];   // [64,1024,500] spikes (0/1)
  const float* W1 = (const float*)d_in[1];   // [1024,1024]
  const float* W2 = (const float*)d_in[2];   // [10,1024]
  float* out = (float*)d_out;                // [64,10,500]

  constexpr int Bb = 64, NIN = 1024, NHID = 1024, NOUT = 10, NOUTP = 16;
  constexpr int T = T_N, Tp = T_PAD;

  // ---- workspace bump allocator (256B aligned) ----
  char* p = (char*)d_ws;
  auto alloc = [&](size_t bytes) { void* r = (void*)p; p += (bytes + 255) & ~(size_t)255; return r; };
  __bf16* xbf  = (__bf16*)alloc((size_t)Bb * NIN * Tp * 2);   // reused as S1bf
  float*  U1   = (float*) alloc((size_t)Bb * NHID * T * 4);
  float*  Y1   = (float*) alloc((size_t)Bb * NHID * T * 4);
  __bf16* W1h  = (__bf16*)alloc((size_t)NHID * NIN * 2);
  __bf16* W1l  = (__bf16*)alloc((size_t)NHID * NIN * 2);
  __bf16* W2h  = (__bf16*)alloc((size_t)NOUTP * NHID * 2);
  __bf16* W2l  = (__bf16*)alloc((size_t)NOUTP * NHID * 2);
  float*  U2   = (float*) alloc((size_t)Bb * NOUT * T * 4);
  float*  Y2   = (float*) alloc((size_t)Bb * NOUT * T * 4);
  __bf16* S1bf = xbf;   // xbf dead after GEMM1; same size/layout

  // ---- one-time packing ----
  {
    int n1 = NHID * NIN;
    snn_pack_w<<<(n1 + 255) / 256, 256, 0, stream>>>(W1, W1h, W1l, NHID, NHID, NIN);
    int n2 = NOUTP * NHID;
    snn_pack_w<<<(n2 + 255) / 256, 256, 0, stream>>>(W2, W2h, W2l, NOUT, NOUTP, NHID);
    long nx = (long)Bb * NIN * Tp;
    snn_pack_x<<<(unsigned)((nx + 255) / 256), 256, 0, stream>>>(x, xbf, T, Tp, nx);
  }

  // ---- layer 1: dense -> psp -> spike(bf16) ----
  {
    constexpr int WMv = 8, WNv = 1, MBv = WMv * 16, NBv = WNv * 16;
    int numM = NHID / MBv;               // 8
    int numT = (T + NBv - 1) / NBv;      // 32
    dim3 grid(numM * numT, Bb);
    snn_wmma_gemm<WMv, WNv><<<grid, 256, 0, stream>>>(W1h, W1l, xbf, U1, NHID, NIN, T, Tp, numM);
  }
  snn_psp<<<Bb * NHID, 512, 0, stream>>>(U1, Y1, T);
  snn_spike<__bf16><<<(Bb * NHID + 255) / 256, 256, 0, stream>>>(Y1, S1bf, Bb * NHID, T, Tp);

  // ---- layer 2: dense -> psp -> spike(f32 -> d_out) ----
  {
    constexpr int WMv = 1, WNv = 8, MBv = WMv * 16, NBv = WNv * 16;
    int numM = 1;
    int numT = (T + NBv - 1) / NBv;      // 4
    dim3 grid(numM * numT, Bb);
    snn_wmma_gemm<WMv, WNv><<<grid, 256, 0, stream>>>(W2h, W2l, S1bf, U2, NOUT, NHID, T, Tp, numM);
  }
  snn_psp<<<Bb * NOUT, 512, 0, stream>>>(U2, Y2, T);
  snn_spike<float><<<(Bb * NOUT + 255) / 256, 256, 0, stream>>>(Y2, out, Bb * NOUT, T, T);
}
